// DepthNet_24300924961372
// MI455X (gfx1250) — matrix operand order repaired and checked by reading, not compiled, not run on hardware
//
#include <hip/hip_runtime.h>

// ---------------- constants ----------------
#define HH 192
#define WW 192
#define HWPIX (HH * WW)
#define AN_ 7
#define AN2_ 49
#define NS_ 4

typedef __attribute__((ext_vector_type(16))) _Float16 v16h;
typedef __attribute__((ext_vector_type(8)))  _Float16 v8h;
typedef __attribute__((ext_vector_type(8)))  float    v8f;

static inline int ru32i(int x) { return (x + 31) & ~31; }
static inline int ru16i(int x) { return (x + 15) & ~15; }

// ---------------- device helpers ----------------
__device__ __forceinline__ float bilin(const float* __restrict__ img, float py, float px) {
    float y0f = floorf(py), x0f = floorf(px);
    int y0 = (int)y0f, x0 = (int)x0f;
    float wy = py - y0f, wx = px - x0f;
    float v00 = 0.f, v01 = 0.f, v10 = 0.f, v11 = 0.f;
    bool y0i = (y0 >= 0) && (y0 <= HH - 1);
    bool y1i = (y0 + 1 >= 0) && (y0 + 1 <= HH - 1);
    bool x0i = (x0 >= 0) && (x0 <= WW - 1);
    bool x1i = (x0 + 1 >= 0) && (x0 + 1 <= WW - 1);
    if (y0i && x0i) v00 = img[y0 * WW + x0];
    if (y0i && x1i) v01 = img[y0 * WW + x0 + 1];
    if (y1i && x0i) v10 = img[(y0 + 1) * WW + x0];
    if (y1i && x1i) v11 = img[(y0 + 1) * WW + x0 + 1];
    return v00 * (1.f - wy) * (1.f - wx) + v01 * (1.f - wy) * wx
         + v10 * wy * (1.f - wx)        + v11 * wy * wx;
}

__device__ __forceinline__ v16h make_afrag(const _Float16* aLds, int kb, int b0) {
    v8h a0 = *(const v8h*)(aLds + kb + b0);
    v8h a1 = *(const v8h*)(aLds + kb + b0 + 8);
    v16h a;
#pragma unroll
    for (int i = 0; i < 8; ++i) { a[i] = a0[i]; a[i + 8] = a1[i]; }
    return a;
}

// ---------------- WMMA GEMM ----------------
// D[n][m] = sum_k A[n][k] * B[m][k]   (A: f16 [Npad16][Kpad], B: f16 col [M][Kpad])
// The 16-row A panel for this block's n-tile is staged into LDS ONCE
// (<= 16*2880*2B = 92 KB, fits the 320 KB WGP LDS); the k-loop runs barrier-free.
// Each wave computes FOUR adjacent 16x16 output tiles sharing one A fragment,
// with the next k-step's four B fragments double-buffered in registers so the
// WMMAs issue while the next 8 global_load_b128 remain outstanding (no
// s_wait_loadcnt 0 drains inside the steady state).
// act: 0=none, 1=LeakyReLU(0.1), 2=ReLU. Output fp32 (outF) or f16 (outH),
// element (n, pix) at n*outChanStride + pix.
__global__ __launch_bounds__(256)
void wmma_gemm_k(const _Float16* __restrict__ A,
                 const _Float16* __restrict__ B,
                 const float* __restrict__ bias,
                 float* __restrict__ outF, _Float16* __restrict__ outH,
                 int N, int Kpad, long outChanStride, int act) {
    extern __shared__ _Float16 AsDyn[];       // [16][Kpad], row-major
    const int lane = threadIdx.x & 31;
    const int wave = threadIdx.x >> 5;
    const int nBase = blockIdx.y * 16;

    // cooperative one-shot staging of the A panel (16B chunks)
    const int panelHalves = 16 * Kpad;        // Kpad % 32 == 0 -> divisible by 8
    const _Float16* aPanel = A + (size_t)nBase * Kpad;
    for (int i = threadIdx.x * 8; i < panelHalves; i += 256 * 8)
        *(v8h*)(&AsDyn[i]) = *(const v8h*)(aPanel + i);
    __syncthreads();

    // fragment addressing per CDNA5 WMMA layouts
    const int pixBase = (blockIdx.x * 8 + wave) * 64;   // four 16-pixel tiles per wave
    const int rowSel = lane & 15;
    const int b0 = (lane < 16) ? 0 : 16;                // A: K sub-block select
    const _Float16* aLds = AsDyn + (size_t)rowSel * Kpad;
    const _Float16* bG0 = B + (size_t)(pixBase + rowSel) * Kpad + ((lane < 16) ? 0 : 16);
    const size_t bT = (size_t)16 * Kpad;

    v8f acc0 = {}, acc1 = {}, acc2 = {}, acc3 = {};

    // prologue: current B fragments for kb = 0
    v16h cb0 = *(const v16h*)(bG0);
    v16h cb1 = *(const v16h*)(bG0 + bT);
    v16h cb2 = *(const v16h*)(bG0 + 2 * bT);
    v16h cb3 = *(const v16h*)(bG0 + 3 * bT);

    int kb = 0;
    for (; kb < Kpad - 32; kb += 32) {
        // prefetch next step's B fragments (independent registers)
        v16h nb0 = *(const v16h*)(bG0 + kb + 32);
        v16h nb1 = *(const v16h*)(bG0 + bT + kb + 32);
        v16h nb2 = *(const v16h*)(bG0 + 2 * bT + kb + 32);
        v16h nb3 = *(const v16h*)(bG0 + 3 * bT + kb + 32);
        v16h a = make_afrag(aLds, kb, b0);
        acc0 = __builtin_amdgcn_wmma_f32_16x16x32_f16(
            false, a, false, cb0, (short)0, acc0, false, false);
        acc1 = __builtin_amdgcn_wmma_f32_16x16x32_f16(
            false, a, false, cb1, (short)0, acc1, false, false);
        acc2 = __builtin_amdgcn_wmma_f32_16x16x32_f16(
            false, a, false, cb2, (short)0, acc2, false, false);
        acc3 = __builtin_amdgcn_wmma_f32_16x16x32_f16(
            false, a, false, cb3, (short)0, acc3, false, false);
        cb0 = nb0; cb1 = nb1; cb2 = nb2; cb3 = nb3;
    }
    // epilogue: last k-step
    {
        v16h a = make_afrag(aLds, kb, b0);
        acc0 = __builtin_amdgcn_wmma_f32_16x16x32_f16(
            false, a, false, cb0, (short)0, acc0, false, false);
        acc1 = __builtin_amdgcn_wmma_f32_16x16x32_f16(
            false, a, false, cb1, (short)0, acc1, false, false);
        acc2 = __builtin_amdgcn_wmma_f32_16x16x32_f16(
            false, a, false, cb2, (short)0, acc2, false, false);
        acc3 = __builtin_amdgcn_wmma_f32_16x16x32_f16(
            false, a, false, cb3, (short)0, acc3, false, false);
    }

    const int nOff = (lane < 16) ? 0 : 8;     // C/D layout: VGPR v -> M row v (+8 hi lanes)
    const int pix = pixBase + (lane & 15);
#pragma unroll
    for (int v = 0; v < 8; ++v) {
        int n = nBase + nOff + v;
        if (n < N) {
            float bv = bias[n];
            float o0 = acc0[v] + bv;
            float o1 = acc1[v] + bv;
            float o2 = acc2[v] + bv;
            float o3 = acc3[v] + bv;
            if (act == 1) {
                o0 = (o0 >= 0.f) ? o0 : 0.1f * o0;
                o1 = (o1 >= 0.f) ? o1 : 0.1f * o1;
                o2 = (o2 >= 0.f) ? o2 : 0.1f * o2;
                o3 = (o3 >= 0.f) ? o3 : 0.1f * o3;
            } else if (act == 2) {
                o0 = fmaxf(o0, 0.f); o1 = fmaxf(o1, 0.f);
                o2 = fmaxf(o2, 0.f); o3 = fmaxf(o3, 0.f);
            }
            size_t o = (size_t)n * (size_t)outChanStride + (size_t)pix;
            if (outF) {
                outF[o] = o0; outF[o + 16] = o1; outF[o + 32] = o2; outF[o + 48] = o3;
            } else {
                outH[o] = (_Float16)o0;      outH[o + 16] = (_Float16)o1;
                outH[o + 32] = (_Float16)o2; outH[o + 48] = (_Float16)o3;
            }
        }
    }
}

// ---------------- weight conversion fp32 -> f16 [Npad][Kpad] ----------------
__global__ void convert_w_k(const float* __restrict__ w, _Float16* __restrict__ A,
                            int N, int K, int Kpad) {
    int i = blockIdx.x * 256 + threadIdx.x;
    if (i >= N * K) return;
    int n = i / K, k = i - n * K;
    A[(size_t)n * Kpad + k] = (_Float16)w[i];
}

// ---------------- regular im2col (zero pad) -> col[M][Kpad] f16 ----------------
__global__ void im2col_k(const float* __restrict__ x, _Float16* __restrict__ col,
                         int ksz, int pad, int K, int Kpad) {
    int m = blockIdx.x * 256 + threadIdx.x;
    int cin = blockIdx.y;
    int yy = m / WW, xx = m - yy * WW;
    int py0 = yy - pad, px0 = xx - pad;
    const float* xc = x + (size_t)cin * HWPIX;
    _Float16* cm = col + (size_t)m * Kpad + (size_t)cin * ksz * ksz;
    for (int ky = 0; ky < ksz; ++ky)
        for (int kx = 0; kx < ksz; ++kx) {
            int y = py0 + ky, xcoord = px0 + kx;
            float v = 0.f;
            if (y >= 0 && y < HH && xcoord >= 0 && xcoord < WW) v = xc[y * WW + xcoord];
            cm[ky * ksz + kx] = (_Float16)v;
        }
    if (cin == 0)
        for (int k = K; k < Kpad; ++k) col[(size_t)m * Kpad + k] = (_Float16)0.f;
}

// ------------- deformable sampling * modulation -> col[M][Kpad] f16 -------------
// offmod fp32: rows [0..2K2): offsets (p*2+{y,x}); rows [2K2..3K2): raw mod
__global__ void dsample_k(const float* __restrict__ x, const float* __restrict__ offmod,
                          _Float16* __restrict__ col, int ksz, int pad, int K, int Kpad) {
    int m = blockIdx.x * 256 + threadIdx.x;
    int cin = blockIdx.y;
    int K2 = ksz * ksz;
    int yy = m / WW, xx = m - yy * WW;
    const float* xc = x + (size_t)cin * HWPIX;
    _Float16* cm = col + (size_t)m * Kpad + (size_t)cin * K2;
    for (int p = 0; p < K2; ++p) {
        float offY = offmod[(size_t)(2 * p) * HWPIX + m];
        float offX = offmod[(size_t)(2 * p + 1) * HWPIX + m];
        float raw = offmod[(size_t)(2 * K2 + p) * HWPIX + m];
        float modv = 2.f / (1.f + expf(-raw));
        float py = (float)(yy - pad + p / ksz) + offY;
        float px = (float)(xx - pad + p % ksz) + offX;
        cm[p] = (_Float16)(bilin(xc, py, px) * modv);
    }
    if (cin == 0)
        for (int k = K; k < Kpad; ++k) col[(size_t)m * Kpad + k] = (_Float16)0.f;
}

// ---------------- disparity warping ----------------
__global__ void warp_k(const float* __restrict__ img, const float* __restrict__ disp,
                       const int* __restrict__ ind, float* __restrict__ warped) {
    int m = blockIdx.x * 256 + threadIdx.x;
    int s = blockIdx.y;   // 0..3 source views
    int t = blockIdx.z;   // 0..48 target views
    int sv = ind[s];
    float dw = (float)(t % AN_ - sv % AN_);
    float dh = (float)(t / AN_ - sv / AN_);
    float d = disp[(size_t)t * HWPIX + m];
    float gy = (float)(m / WW) + d * dh;
    float gx = (float)(m % WW) + d * dw;
    warped[((size_t)t * NS_ + s) * HWPIX + m] = bilin(img + (size_t)s * HWPIX, gy, gx);
}

// ---------------- im2col for 3D conv (f16 feat [Cin][D][HW]) ----------------
__global__ void im2col3d_k(const _Float16* __restrict__ feat, _Float16* __restrict__ col,
                           int D, int kd, int d0, int K, int Kpad) {
    int m = blockIdx.x * 256 + threadIdx.x;
    int cin = blockIdx.y;
    int yy = m / WW, xx = m - yy * WW;
    int y0 = yy - 1, x0 = xx - 1;
    _Float16* cm = col + (size_t)m * Kpad + (size_t)cin * kd * 9;
    for (int dz = 0; dz < kd; ++dz)
        for (int ky = 0; ky < 3; ++ky)
            for (int kx = 0; kx < 3; ++kx) {
                int y = y0 + ky, xcoord = x0 + kx;
                _Float16 v = (_Float16)0.f;
                if (y >= 0 && y < HH && xcoord >= 0 && xcoord < WW)
                    v = feat[((size_t)cin * D + (d0 + dz)) * HWPIX + y * WW + xcoord];
                cm[dz * 9 + ky * 3 + kx] = v;
            }
    if (cin == 0)
        for (int k = K; k < Kpad; ++k) col[(size_t)m * Kpad + k] = (_Float16)0.f;
}

// ---------------- final: out = warped[:, 0] + residual ----------------
__global__ void final_k(const float* __restrict__ warped, const float* __restrict__ r3,
                        float* __restrict__ out) {
    int i = blockIdx.x * 256 + threadIdx.x;
    int t = i / HWPIX, m = i - t * HWPIX;
    out[i] = warped[(size_t)t * NS_ * HWPIX + m] + r3[i];
}

// ---------------- host helpers ----------------
struct Scratch {
    _Float16* colBuf;   // max 36864 x 2880 f16
    float* offmod;      // 160 x HW fp32
    _Float16* AbufOff;  // offsets/mod weights, f16 padded
    _Float16* AbufMain; // main weights, f16 padded
    float* biasOff;     // 160 fp32
    float* biasMain;    // 64 fp32
};

static void launch_gemm(hipStream_t stream, const Scratch& S,
                        const _Float16* A, const float* bias,
                        float* outF, _Float16* outH,
                        int N, int Kpad, long outChanStride, int act) {
    size_t shmem = (size_t)16 * Kpad * sizeof(_Float16);  // <= 92 KB (WGP LDS 320 KB)
    // 8 waves/block, 64 pixels per wave -> 512 pixels per block
    wmma_gemm_k<<<dim3(HWPIX / 512, (N + 15) / 16), 256, shmem, stream>>>(
        A, S.colBuf, bias, outF, outH, N, Kpad, outChanStride, act);
}

static void run_deform(hipStream_t stream, void* const* d_in, int base,
                       const float* x, int Cin, int Cout, int ksz, int pad,
                       float* outF, _Float16* outH, long outChanStride, int act,
                       const Scratch& S, bool doConvert) {
    int K2 = ksz * ksz;
    int K = Cin * K2;
    int Kpad = ru32i(K);
    int Noff = 3 * K2;
    int NoffPad = ru16i(Noff);
    int NPad = ru16i(Cout);
    const float* w  = (const float*)d_in[base + 0];
    const float* bb = (const float*)d_in[base + 1];
    const float* ow = (const float*)d_in[base + 2];
    const float* ob = (const float*)d_in[base + 3];
    const float* mw = (const float*)d_in[base + 4];
    const float* mb = (const float*)d_in[base + 5];

    if (doConvert) {
        hipMemsetAsync(S.AbufOff, 0, (size_t)NoffPad * Kpad * sizeof(_Float16), stream);
        int tot = 2 * K2 * K;
        convert_w_k<<<(tot + 255) / 256, 256, 0, stream>>>(ow, S.AbufOff, 2 * K2, K, Kpad);
        tot = K2 * K;
        convert_w_k<<<(tot + 255) / 256, 256, 0, stream>>>(
            mw, S.AbufOff + (size_t)2 * K2 * Kpad, K2, K, Kpad);
        hipMemcpyAsync(S.biasOff, ob, (size_t)2 * K2 * sizeof(float),
                       hipMemcpyDeviceToDevice, stream);
        hipMemcpyAsync(S.biasOff + 2 * K2, mb, (size_t)K2 * sizeof(float),
                       hipMemcpyDeviceToDevice, stream);
        hipMemsetAsync(S.AbufMain, 0, (size_t)NPad * Kpad * sizeof(_Float16), stream);
        tot = Cout * K;
        convert_w_k<<<(tot + 255) / 256, 256, 0, stream>>>(w, S.AbufMain, Cout, K, Kpad);
        hipMemcpyAsync(S.biasMain, bb, (size_t)Cout * sizeof(float),
                       hipMemcpyDeviceToDevice, stream);
    }
    // offsets + modulation conv (regular conv as GEMM)
    im2col_k<<<dim3(HWPIX / 256, Cin), 256, 0, stream>>>(x, S.colBuf, ksz, pad, K, Kpad);
    launch_gemm(stream, S, S.AbufOff, S.biasOff, S.offmod, nullptr,
                Noff, Kpad, (long)HWPIX, 0);
    // deformable sampling -> column buffer, then main GEMM
    dsample_k<<<dim3(HWPIX / 256, Cin), 256, 0, stream>>>(x, S.offmod, S.colBuf,
                                                          ksz, pad, K, Kpad);
    launch_gemm(stream, S, S.AbufMain, S.biasMain, outF, outH,
                Cout, Kpad, outChanStride, act);
}

static void run_conv3d(hipStream_t stream, const float* w, const float* b,
                       const _Float16* feat, int Cin, int Cout, int D, int kd,
                       int strideD, int Dout,
                       float* outF, _Float16* outH, long outChanStride, int act,
                       const Scratch& S) {
    int K = Cin * kd * 9;
    int Kpad = ru32i(K);
    int NPad = ru16i(Cout);
    hipMemsetAsync(S.AbufMain, 0, (size_t)NPad * Kpad * sizeof(_Float16), stream);
    int tot = Cout * K;
    convert_w_k<<<(tot + 255) / 256, 256, 0, stream>>>(w, S.AbufMain, Cout, K, Kpad);
    hipMemcpyAsync(S.biasMain, b, (size_t)Cout * sizeof(float),
                   hipMemcpyDeviceToDevice, stream);
    for (int dz = 0; dz < Dout; ++dz) {
        im2col3d_k<<<dim3(HWPIX / 256, Cin), 256, 0, stream>>>(
            feat, S.colBuf, D, kd, dz * strideD, K, Kpad);
        launch_gemm(stream, S, S.AbufMain, S.biasMain,
                    outF ? outF + (size_t)dz * HWPIX : nullptr,
                    outH ? outH + (size_t)dz * HWPIX : nullptr,
                    Cout, Kpad, outChanStride, act);
    }
}

// ---------------- entry ----------------
// d_in order: 0 ind_source, 1 img_source, 2 LFI,
//   3..8 conv1{w,b,ow,ob,mw,mb}, 9..14 conv2, 15..20 x1, 21..26 x2, 27..32 x3,
//   33..38 x4, 39..44 x5, 45..50 confuse, 51..56 lf_conv0,
//   57 c3d1w, 58 c3d1b, 59 c3d2w, 60 c3d2b, 61 c3d3w, 62 c3d3b
extern "C" void kernel_launch(void* const* d_in, const int* in_sizes, int n_in,
                              void* d_out, int out_size, void* d_ws, size_t ws_size,
                              hipStream_t stream) {
    (void)in_sizes; (void)n_in; (void)out_size; (void)ws_size;
    const int* ind = (const int*)d_in[0];
    const float* img = (const float*)d_in[1];

    // deterministic bump allocator over d_ws
    char* ws = (char*)d_ws;
    size_t off = 0;
    auto alloc = [&](size_t bytes) -> void* {
        void* p = ws + off;
        off = (off + bytes + 255) & ~(size_t)255;
        return p;
    };
    Scratch S;
    S.colBuf   = (_Float16*)alloc((size_t)HWPIX * 2880 * sizeof(_Float16)); // 212 MB
    S.offmod   = (float*)alloc((size_t)160 * HWPIX * sizeof(float));
    S.AbufOff  = (_Float16*)alloc((size_t)160 * 1600 * sizeof(_Float16));
    S.AbufMain = (_Float16*)alloc((size_t)64 * 2880 * sizeof(_Float16));
    S.biasOff  = (float*)alloc(160 * sizeof(float));
    S.biasMain = (float*)alloc(64 * sizeof(float));
    float* catBuf = (float*)alloc((size_t)147 * HWPIX * sizeof(float)); // x_m1|x_m2|x5
    float* x1f = (float*)alloc((size_t)16 * HWPIX * sizeof(float));
    float* x2f = (float*)alloc((size_t)32 * HWPIX * sizeof(float));
    float* x3f = (float*)alloc((size_t)64 * HWPIX * sizeof(float));
    float* x4f = (float*)alloc((size_t)49 * HWPIX * sizeof(float));
    float* disp = (float*)alloc((size_t)AN2_ * HWPIX * sizeof(float));
    float* warped = (float*)alloc((size_t)AN2_ * NS_ * HWPIX * sizeof(float));
    _Float16* depfeat = (_Float16*)alloc((size_t)64 * AN2_ * HWPIX * sizeof(_Float16));
    _Float16* r1 = (_Float16*)alloc((size_t)64 * 12 * HWPIX * sizeof(_Float16));
    _Float16* r2 = (_Float16*)alloc((size_t)64 * 3 * HWPIX * sizeof(_Float16));
    float* r3 = (float*)alloc((size_t)AN2_ * HWPIX * sizeof(float));

    // deformable feature stack (LeakyReLU 0.1 -> act=1)
    run_deform(stream, d_in, 3,  img, NS_, AN2_, 3, 1, catBuf,                nullptr, HWPIX, 1, S, true);
    run_deform(stream, d_in, 9,  img, NS_, AN2_, 5, 2, catBuf + (size_t)49 * HWPIX, nullptr, HWPIX, 1, S, true);
    run_deform(stream, d_in, 15, img, NS_, 16, 3, 1, x1f, nullptr, HWPIX, 1, S, true);
    run_deform(stream, d_in, 21, x1f, 16, 32, 5, 2, x2f, nullptr, HWPIX, 1, S, true);
    run_deform(stream, d_in, 27, x2f, 32, 64, 7, 3, x3f, nullptr, HWPIX, 1, S, true);
    run_deform(stream, d_in, 33, x3f, 64, AN2_, 5, 2, x4f, nullptr, HWPIX, 1, S, true);
    run_deform(stream, d_in, 39, x4f, AN2_, AN2_, 3, 1, catBuf + (size_t)98 * HWPIX, nullptr, HWPIX, 1, S, true);
    // confuse: 1x1 deform conv over concat -> disparity (no activation)
    run_deform(stream, d_in, 45, catBuf, 147, AN2_, 1, 0, disp, nullptr, HWPIX, 0, S, true);

    // warp the 4 source views to all 49 target views
    warp_k<<<dim3(HWPIX / 256, NS_, AN2_), 256, 0, stream>>>(img, disp, ind, warped);

    // lf_conv0 on each warped view (weights converted once) -> depfeat f16 [64][49][HW]
    for (int t = 0; t < AN2_; ++t)
        run_deform(stream, d_in, 51, warped + (size_t)t * NS_ * HWPIX, NS_, 64, 3, 1,
                   nullptr, depfeat + (size_t)t * HWPIX, (long)AN2_ * HWPIX, 1, S, t == 0);

    // 3D conv residual stack (ReLU -> act=2; final layer linear)
    run_conv3d(stream, (const float*)d_in[57], (const float*)d_in[58], depfeat,
               64, 64, AN2_, 5, 4, 12, nullptr, r1, (long)12 * HWPIX, 2, S);
    run_conv3d(stream, (const float*)d_in[59], (const float*)d_in[60], r1,
               64, 64, 12, 4, 4, 3, nullptr, r2, (long)3 * HWPIX, 2, S);
    run_conv3d(stream, (const float*)d_in[61], (const float*)d_in[62], r2,
               64, AN2_, 3, 3, 1, 1, r3, nullptr, (long)HWPIX, 0, S);

    final_k<<<(AN2_ * HWPIX) / 256, 256, 0, stream>>>(warped, r3, (float*)d_out);
}
